// HybridNATGraphQNN_65481071399142
// MI455X (gfx1250) — matrix operand order repaired and checked by reading, not compiled, not run on hardware
//
#include <hip/hip_runtime.h>
#include <hip/hip_bf16.h>

#define BATCH 8192

typedef __attribute__((ext_vector_type(16))) _Float16 v16h;
typedef __attribute__((ext_vector_type(8)))  _Float16 h8;
typedef __attribute__((ext_vector_type(8)))  float    v8f;

// ------------- prep: reorder conv2_w -> Wt[16][96] f16 (K = tap*8+ic, zero-padded to 96),
//               and convert fc1_w -> Wf[64][784] f16 --------------------------------------
__global__ void k_prep_weights(const float* __restrict__ c2w,   // (16,8,3,3)
                               const float* __restrict__ f1w,   // (64,784)
                               _Float16* __restrict__ Wt,       // (16,96)
                               _Float16* __restrict__ Wf)       // (64,784)
{
    int i = blockIdx.x * blockDim.x + threadIdx.x;
    if (i < 16 * 96) {
        int oc = i / 96, k = i % 96;
        float v = 0.0f;
        if (k < 72) {
            int tap = k >> 3, ic = k & 7;                 // k = tap*8 + ic
            v = c2w[oc * 72 + ic * 9 + tap];              // w[oc][ic][ky][kx]
        }
        Wt[i] = (_Float16)v;
    }
    if (i < 64 * 784) Wf[i] = (_Float16)f1w[i];
}

// ------------- conv1 (1->8, 3x3 SAME) + relu + 2x2 maxpool, output NHWC f16 ----------------
// Ah[((n*14+y)*14+x)*8 + ic]
__global__ void k_conv1_relu_pool(const float* __restrict__ x,
                                  const float* __restrict__ w,
                                  const float* __restrict__ bias,
                                  _Float16* __restrict__ Ah)
{
    int idx = blockIdx.x * blockDim.x + threadIdx.x;
    const int total = BATCH * 196 * 8;
    if (idx >= total) return;
    int oc = idx & 7; int t = idx >> 3;
    int px = t % 14;  t /= 14;
    int py = t % 14;  int n = t / 14;

    const float* __restrict__ xin = x + (size_t)n * 784;
    const float* __restrict__ wf  = w + oc * 9;
    float bb = bias[oc];
    float m = 0.0f;                               // relu outputs >= 0
#pragma unroll
    for (int dy = 0; dy < 2; ++dy) {
#pragma unroll
        for (int dx = 0; dx < 2; ++dx) {
            int cy = py * 2 + dy, cx = px * 2 + dx;
            float acc = bb;
#pragma unroll
            for (int ky = 0; ky < 3; ++ky) {
                int iy = cy + ky - 1;
                if ((unsigned)iy < 28u) {
#pragma unroll
                    for (int kx = 0; kx < 3; ++kx) {
                        int ix = cx + kx - 1;
                        if ((unsigned)ix < 28u)
                            acc = fmaf(xin[iy * 28 + ix], wf[ky * 3 + kx], acc);
                    }
                }
            }
            acc = acc > 0.f ? acc : 0.f;
            m = acc > m ? acc : m;
        }
    }
    Ah[idx] = (_Float16)m;
}

// ------------- conv2 as implicit GEMM via WMMA: (B*196 x 72) @ (72 x 16) -------------------
// A row = pre-pool spatial position; K = tap*8+ic -> one h8 chunk = one shifted NHWC pixel.
__device__ __forceinline__ h8 tap_chunk(const _Float16* __restrict__ Ah,
                                        int n, int cy, int cx, int tap)
{
    h8 z = {};
    if (tap >= 9) return z;
    int iy = cy + tap / 3 - 1;
    int ix = cx + tap % 3 - 1;
    if ((unsigned)iy >= 14u || (unsigned)ix >= 14u) return z;
    return *(const h8*)(Ah + (((size_t)n * 14 + iy) * 14 + ix) * 8);
}

__global__ void __launch_bounds__(128)
k_conv2_wmma(const _Float16* __restrict__ Ah,   // [B][14][14][8] f16
             const _Float16* __restrict__ Wt,   // [16][96] f16 (zero-padded K)
             const float* __restrict__ bias,    // 16
             _Float16* __restrict__ P)          // [B*196][16] f16, relu'd pre-pool
{
    const int lane    = threadIdx.x & 31;
    const int wave    = threadIdx.x >> 5;
    const int tileRow = (blockIdx.x * 4 + wave) * 16;
    const int oc      = lane & 15;
    const int row     = tileRow + (lane & 15);
    const int n       = row / 196;
    const int pos     = row % 196;
    const int cy      = pos / 14, cx = pos % 14;
    const int tsel    = lane >> 4;                       // 0 or 1
    const _Float16* __restrict__ wrow = Wt + oc * 96 + tsel * 8;

    v8f acc = {};
#pragma unroll
    for (int s = 0; s < 3; ++s) {
        // A: lane<16 -> taps 4s, 4s+2 ; lane>=16 -> taps 4s+1, 4s+3 (>=9 -> zeros)
        int tapA = 4 * s + tsel;
        h8 c0 = tap_chunk(Ah, n, cy, cx, tapA);
        h8 c1 = tap_chunk(Ah, n, cy, cx, tapA + 2);
        v16h afrag = __builtin_shufflevector(c0, c1, 0,1,2,3,4,5,6,7,8,9,10,11,12,13,14,15);
        // B: K chunks at 32s + 8*tsel and 32s + 16 + 8*tsel (padded, always valid)
        h8 b0 = *(const h8*)(wrow + 32 * s);
        h8 b1 = *(const h8*)(wrow + 32 * s + 16);
        v16h bfrag = __builtin_shufflevector(b0, b1, 0,1,2,3,4,5,6,7,8,9,10,11,12,13,14,15);
        acc = __builtin_amdgcn_wmma_f32_16x16x32_f16(
                  false, afrag, false, bfrag, (short)0, acc, false, false);
    }

    const int rbase = (lane >> 4) * 8;
    const float bb = bias[oc];
#pragma unroll
    for (int r = 0; r < 8; ++r) {
        int orow = tileRow + rbase + r;
        float v = acc[r] + bb;
        v = v > 0.f ? v : 0.f;
        P[(size_t)orow * 16 + oc] = (_Float16)v;
    }
}

// ------------- 2x2 maxpool on P -> Bh in flatten-of-(B,16,7,7) order (f16) -----------------
__global__ void k_pool2(const _Float16* __restrict__ P,
                        _Float16* __restrict__ Bh)
{
    int idx = blockIdx.x * blockDim.x + threadIdx.x;
    if (idx >= BATCH * 784) return;
    int px = idx % 7; int t = idx / 7;
    int py = t % 7;   t /= 7;
    int oc = t % 16;  int n = t / 16;
    const size_t base = (size_t)n * 196;
    float m = 0.0f;
#pragma unroll
    for (int dy = 0; dy < 2; ++dy)
#pragma unroll
        for (int dx = 0; dx < 2; ++dx) {
            float v = (float)P[(base + (2 * py + dy) * 14 + (2 * px + dx)) * 16 + oc];
            m = v > m ? v : m;
        }
    Bh[idx] = (_Float16)m;
}

// ------------- fc1: (8192x784) @ (784x64) + bias, relu — pure f16 WMMA ---------------------
__global__ void __launch_bounds__(128)
k_fc1_wmma(const _Float16* __restrict__ Bh,   // 8192 x 784 f16
           const _Float16* __restrict__ Wf,   // 64 x 784 f16
           const float* __restrict__ bias,    // 64
           float* __restrict__ C)             // 8192 x 64
{
    const int lane  = threadIdx.x & 31;
    const int wave  = threadIdx.x >> 5;           // 0..3 -> N tile
    const int tileM = blockIdx.x * 16;
    const int nCol  = wave * 16 + (lane & 15);
    const int row   = tileM + (lane & 15);
    const int kA0   = (lane >> 4) * 8;            // 0 or 8
    const int kB0   = (lane >> 4) * 8;            // 0 or 8

    const _Float16* __restrict__ arow = Bh + (size_t)row * 784;
    const _Float16* __restrict__ wrow = Wf + (size_t)nCol * 784;

    v8f acc = {};
    for (int kk = 0; kk < 768; kk += 32) {
        const int ka = kk + kA0;
        h8 a0 = *(const h8*)(arow + ka);
        h8 a1 = *(const h8*)(arow + ka + 16);
        v16h afrag = __builtin_shufflevector(a0, a1, 0,1,2,3,4,5,6,7,8,9,10,11,12,13,14,15);
        const int kb = kk + kB0;
        h8 b0 = *(const h8*)(wrow + kb);
        h8 b1 = *(const h8*)(wrow + kb + 16);
        v16h bfrag = __builtin_shufflevector(b0, b1, 0,1,2,3,4,5,6,7,8,9,10,11,12,13,14,15);
        acc = __builtin_amdgcn_wmma_f32_16x16x32_f16(
                  false, afrag, false, bfrag, (short)0, acc, false, false);
    }
    { // tail: K=768..783 valid; upper A and B chunks are zero
        h8 z = {};
        h8 a0 = *(const h8*)(arow + 768 + kA0);
        v16h afrag = __builtin_shufflevector(a0, z, 0,1,2,3,4,5,6,7,8,9,10,11,12,13,14,15);
        h8 b0 = *(const h8*)(wrow + 768 + kA0), b1 = z;
        v16h bfrag = __builtin_shufflevector(b0, b1, 0,1,2,3,4,5,6,7,8,9,10,11,12,13,14,15);
        acc = __builtin_amdgcn_wmma_f32_16x16x32_f16(
                  false, afrag, false, bfrag, (short)0, acc, false, false);
    }

    const int rbase = (lane >> 4) * 8;
    const float bn = bias[nCol];
#pragma unroll
    for (int r = 0; r < 8; ++r) {
        int orow = tileM + rbase + r;
        float v = acc[r] + bn;
        C[(size_t)orow * 64 + nCol] = v > 0.f ? v : 0.f;
    }
}

// ------------- fc2: (8192x64) @ (64x8) + bias -> qs ----------------------------------------
__global__ void k_fc2(const float* __restrict__ C,
                      const float* __restrict__ W,
                      const float* __restrict__ bias,
                      float* __restrict__ qs)
{
    int idx = blockIdx.x * blockDim.x + threadIdx.x;
    if (idx >= BATCH * 8) return;
    int nf = idx & 7; int row = idx >> 3;
    const float* __restrict__ c = C + (size_t)row * 64;
    const float* __restrict__ w = W + nf * 64;
    float acc = bias[nf];
#pragma unroll
    for (int k = 0; k < 64; ++k) acc = fmaf(c[k], w[k], acc);
    qs[idx] = acc;
}

// ------------- batchnorm statistics over the batch (8 features) ----------------------------
__global__ void k_bn_stats(const float* __restrict__ qs,
                           const float* __restrict__ gamma,
                           const float* __restrict__ beta,
                           float* __restrict__ stats)   // [0..7]=scale, [8..15]=shift
{
    __shared__ float ssum[256];
    __shared__ float ssq[256];
    const int f   = blockIdx.x;
    const int tid = threadIdx.x;
    float s = 0.f, q = 0.f;
    for (int i = tid; i < BATCH; i += 256) {
        float v = qs[(size_t)i * 8 + f];
        s += v;
        q = fmaf(v, v, q);
    }
    ssum[tid] = s; ssq[tid] = q;
    __syncthreads();
    for (int off = 128; off > 0; off >>= 1) {
        if (tid < off) { ssum[tid] += ssum[tid + off]; ssq[tid] += ssq[tid + off]; }
        __syncthreads();
    }
    if (tid == 0) {
        float mean = ssum[0] * (1.0f / BATCH);
        float var  = ssq[0] * (1.0f / BATCH) - mean * mean;
        float sc   = gamma[f] * rsqrtf(var + 1e-5f);
        stats[f]     = sc;
        stats[8 + f] = beta[f] - mean * sc;
    }
}

// ------------- BN-apply + quantum circuit (analytic collapse) + final head -----------------
// qout[b,w] = prod_{j<=w} cos(qs_norm[b,j] + q_weights[j])
__global__ void k_final(const float* __restrict__ qs,
                        const float* __restrict__ stats,
                        const float* __restrict__ qw,
                        const float* __restrict__ fw,     // 4 x 16
                        const float* __restrict__ fb,     // 4
                        float* __restrict__ out)          // 8192 x 4
{
    int i = blockIdx.x * blockDim.x + threadIdx.x;
    if (i >= BATCH) return;
    float qn[8], qo[8];
    float p = 1.0f;
#pragma unroll
    for (int w = 0; w < 8; ++w) {
        float v = fmaf(qs[(size_t)i * 8 + w], stats[w], stats[8 + w]);
        qn[w] = v;
        p *= cosf(v + qw[w]);
        qo[w] = p;
    }
#pragma unroll
    for (int o = 0; o < 4; ++o) {
        float acc = fb[o];
#pragma unroll
        for (int j = 0; j < 8; ++j) {
            acc = fmaf(fw[o * 16 + j],     qn[j], acc);
            acc = fmaf(fw[o * 16 + 8 + j], qo[j], acc);
        }
        out[(size_t)i * 4 + o] = acc;
    }
}

extern "C" void kernel_launch(void* const* d_in, const int* in_sizes, int n_in,
                              void* d_out, int out_size, void* d_ws, size_t ws_size,
                              hipStream_t stream)
{
    (void)in_sizes; (void)n_in; (void)out_size; (void)ws_size;

    const float* x       = (const float*)d_in[0];
    const float* conv1_w = (const float*)d_in[1];
    const float* conv1_b = (const float*)d_in[2];
    const float* conv2_w = (const float*)d_in[3];
    const float* conv2_b = (const float*)d_in[4];
    const float* fc1_w   = (const float*)d_in[5];
    const float* fc1_b   = (const float*)d_in[6];
    const float* fc2_w   = (const float*)d_in[7];
    const float* fc2_b   = (const float*)d_in[8];
    const float* bn_g    = (const float*)d_in[9];
    const float* bn_b    = (const float*)d_in[10];
    const float* q_w     = (const float*)d_in[11];
    const float* final_w = (const float*)d_in[12];
    const float* final_b = (const float*)d_in[13];
    float* out = (float*)d_out;

    // workspace layout (bytes); everything stays L2-resident (~92 MB < 192 MB L2)
    char* p = (char*)d_ws;
    _Float16* Ah = (_Float16*)p; p += (size_t)BATCH * 196 * 8  * 2;  // conv1 out, NHWC f16
    _Float16* P  = (_Float16*)p; p += (size_t)BATCH * 196 * 16 * 2;  // conv2 pre-pool f16
    _Float16* Bh = (_Float16*)p; p += (size_t)BATCH * 784 * 2;       // pooled, flat f16
    _Float16* Wt = (_Float16*)p; p += (size_t)16 * 96 * 2;           // conv2 W reordered f16
    _Float16* Wf = (_Float16*)p; p += (size_t)64 * 784 * 2;          // fc1 W f16
    float*    C  = (float*)p;    p += (size_t)BATCH * 64 * 4;        // fc1 out
    float*    QS = (float*)p;    p += (size_t)BATCH * 8 * 4;         // fc2 out
    float*    ST = (float*)p;                                        // bn scale/shift (16)

    k_prep_weights<<<(64 * 784 + 255) / 256, 256, 0, stream>>>(conv2_w, fc1_w, Wt, Wf);
    {
        int total = BATCH * 196 * 8;
        k_conv1_relu_pool<<<(total + 255) / 256, 256, 0, stream>>>(x, conv1_w, conv1_b, Ah);
    }
    k_conv2_wmma<<<(BATCH * 196) / 64, 128, 0, stream>>>(Ah, Wt, conv2_b, P);
    {
        int total = BATCH * 784;
        k_pool2<<<(total + 255) / 256, 256, 0, stream>>>(P, Bh);
    }
    k_fc1_wmma<<<BATCH / 16, 128, 0, stream>>>(Bh, Wf, fc1_b, C);
    {
        int total = BATCH * 8;
        k_fc2<<<(total + 255) / 256, 256, 0, stream>>>(C, fc2_w, fc2_b, QS);
    }
    k_bn_stats<<<8, 256, 0, stream>>>(QS, bn_g, bn_b, ST);
    k_final<<<(BATCH + 255) / 256, 256, 0, stream>>>(QS, ST, q_w, final_w, final_b, out);
}
